// Camada_33612414059004
// MI455X (gfx1250) — matrix operand order repaired and checked by reading, not compiled, not run on hardware
//
#include <hip/hip_runtime.h>
#include <hip/hip_bf16.h>
#include <math.h>

// Problem constants (reference: B=64, N=1024, D=8, S=16)
#define BB   64
#define NN   1024
#define DD   8
#define SS   16
#define COLS (DD * SS)     // 128 fused (d,s) columns
#define KC   64            // K-chunk staged in LDS per iteration
#define OT   64            // o-rows per block (4 x 16-row WMMA tiles)
#define LSTRIDE 72         // LDS row stride (bf16): 144B rows, 16B-aligned segments

typedef __attribute__((ext_vector_type(16))) __bf16 v16bf;
typedef __attribute__((ext_vector_type(8)))  __bf16 v8bf;
typedef __attribute__((ext_vector_type(4)))  __bf16 v4bf;
typedef __attribute__((ext_vector_type(2)))  __bf16 v2bf;
typedef __attribute__((ext_vector_type(8)))  float  v8f;
typedef __attribute__((ext_vector_type(4)))  float  v4f;

// ---------------------------------------------------------------------------
// Async global->LDS copy of 16 bytes per lane (gfx1250, tracked by ASYNCcnt).
// LDS destination address from the lds_off VGPR; global addr from 64-bit VGPR pair.
// ---------------------------------------------------------------------------
__device__ __forceinline__ void async_copy_b128(unsigned lds_off,
                                                const void* gaddr) {
    asm volatile("global_load_async_to_lds_b128 %0, %1, off"
                 :: "v"(lds_off), "v"((unsigned long long)gaddr)
                 : "memory");
}
__device__ __forceinline__ void wait_async0() {
    asm volatile("s_wait_asynccnt 0x0" ::: "memory");
}

// ===========================================================================
// Pass 1a: conn [N,N] f32 -> bf16, row-major (exact: entries are 0/1).
// ===========================================================================
__global__ __launch_bounds__(256)
void conn_to_bf16_kernel(const float* __restrict__ conn, __bf16* __restrict__ cb) {
    const size_t i = ((size_t)blockIdx.x * 256 + threadIdx.x) * 8;
    v4f a = *(const v4f*)(conn + i);
    v4f b = *(const v4f*)(conn + i + 4);
    v8bf o;
    #pragma unroll
    for (int j = 0; j < 4; ++j) { o[j] = (__bf16)a[j]; o[j + 4] = (__bf16)b[j]; }
    *(v8bf*)(cb + i) = o;
}

// ===========================================================================
// Pass 1b: x [B,N,D,S] f32 -> xT [B, COLS, N] bf16 (transpose + convert).
// Block: batch blockIdx.y, K-slab blockIdx.x*KC. LDS-tiled for coalescing.
// ===========================================================================
__global__ __launch_bounds__(256)
void x_transpose_bf16_kernel(const float* __restrict__ x, __bf16* __restrict__ xT) {
    __shared__ __bf16 T[COLS * LSTRIDE];
    const int t   = threadIdx.x;
    const int b   = blockIdx.y;
    const int k0  = blockIdx.x * KC;
    const int xr0 = (t >> 5) * 2;      // 0,2,..,14
    const int xc0 = (t & 31) * 4;      // 0..124

    const float* p = x + ((size_t)(b * NN) + k0 + xr0) * COLS + xc0;
    #pragma unroll
    for (int pp = 0; pp < 4; ++pp) {
        v4f v0 = *(const v4f*)(p + (pp * 16) * COLS);
        v4f v1 = *(const v4f*)(p + (pp * 16 + 1) * COLS);
        #pragma unroll
        for (int j = 0; j < 4; ++j) {
            v2bf pk; pk[0] = (__bf16)v0[j]; pk[1] = (__bf16)v1[j];
            *(v2bf*)&T[(xc0 + j) * LSTRIDE + pp * 16 + xr0] = pk;
        }
    }
    __syncthreads();
    // Write coalesced along k: 128 cols x 8 segments of 8 k-values.
    #pragma unroll
    for (int q = 0; q < 4; ++q) {
        const int sgi = t + q * 256;
        const int col = sgi >> 3;
        const int ks  = (sgi & 7) * 8;
        v8bf val = *(const v8bf*)&T[col * LSTRIDE + ks];
        *(v8bf*)(xT + ((size_t)b * COLS + col) * NN + k0 + ks) = val;
    }
}

// ===========================================================================
// Pass 2: fused GEMM + dendrite/soma epilogue, bf16 WMMA, async double-buffer.
// Block: batch blockIdx.y, o-tile blockIdx.x*64. 8 waves; wave w = dendrite d.
// ===========================================================================
__global__ __launch_bounds__(256)
void snn_wmma_async_kernel(const __bf16* __restrict__ xT,     // [B, COLS, N]
                           const __bf16* __restrict__ connb,  // [N, N]
                           const float* __restrict__ w_syn,   // [N, D, S]
                           const float* __restrict__ b_dend,  // [N, D]
                           const float* __restrict__ w_dend,  // [N, D]
                           const float* __restrict__ b_soma,  // [N]
                           float* __restrict__ out)           // [B, N]
{
    __shared__ __bf16 As[2][OT * LSTRIDE];     // conn tile: 64 (o) x KC (k)
    __shared__ __bf16 Bst[2][COLS * LSTRIDE];  // xT tile: 128 (col) x KC (k)
    __shared__ float  partial[OT][DD];

    const int t    = threadIdx.x;
    const int wave = t >> 5;          // dendrite d
    const int lane = t & 31;
    const int half = lane >> 4;
    const int sidx = lane & 15;

    const int b  = blockIdx.y;
    const int o0 = blockIdx.x * OT;

    // Per-lane async-copy coordinates (8 k-elements = 16B per lane).
    const int xcol = wave * 16 + (lane >> 3);     // + q*4 cols per instr
    const int ksub = (lane & 7) * 8;
    const int arow = wave * 8 + (lane >> 3);      // + q*4 rows per instr

    const __bf16* xbase = xT + ((size_t)b * COLS + xcol) * NN + ksub;
    const __bf16* abase = connb + (size_t)(o0 + arow) * NN + ksub;

    v8f acc[4] = {};

    // Issue one K-chunk of async copies into LDS buffer `buf`.
    auto issue_chunk = [&](int k0, int buf) {
        #pragma unroll
        for (int q = 0; q < 4; ++q) {   // X tile: 4 cols per instr per wave
            unsigned loff = (unsigned)(unsigned long long)
                &Bst[buf][(xcol + q * 4) * LSTRIDE + ksub];
            async_copy_b128(loff, xbase + (size_t)(q * 4) * NN + k0);
        }
        #pragma unroll
        for (int q = 0; q < 2; ++q) {   // conn tile: 4 rows per instr per wave
            unsigned loff = (unsigned)(unsigned long long)
                &As[buf][(arow + q * 4) * LSTRIDE + ksub];
            async_copy_b128(loff, abase + (size_t)(q * 4) * NN + k0);
        }
    };

    issue_chunk(0, 0);
    wait_async0();
    __syncthreads();

    int cur = 0;
    for (int k0 = 0; k0 < NN; k0 += KC) {
        const bool more = (k0 + KC < NN);
        if (more) issue_chunk(k0 + KC, cur ^ 1);   // DMA next chunk under WMMA

        #pragma unroll
        for (int kk = 0; kk < KC; kk += 32) {
            // B fragment (32x16): lane = col (wave*16+sidx), 16 contiguous k.
            const __bf16* bp = &Bst[cur][(wave * 16 + sidx) * LSTRIDE + kk + half * 16];
            v8bf b0 = *(const v8bf*)bp;
            v8bf b1 = *(const v8bf*)(bp + 8);
            v16bf bf;
            #pragma unroll
            for (int i = 0; i < 8; ++i) { bf[i] = b0[i]; bf[i + 8] = b1[i]; }

            // All 4 A fragments first (distinct regs -> partial dscnt waits).
            v16bf afr[4];
            #pragma unroll
            for (int mt = 0; mt < 4; ++mt) {
                const __bf16* ap = &As[cur][(mt * 16 + sidx) * LSTRIDE + kk + half * 16];
                v8bf a0 = *(const v8bf*)ap;
                v8bf a1 = *(const v8bf*)(ap + 8);
                #pragma unroll
                for (int i = 0; i < 8; ++i) { afr[mt][i] = a0[i]; afr[mt][i + 8] = a1[i]; }
            }
            #pragma unroll
            for (int mt = 0; mt < 4; ++mt) {
                acc[mt] = __builtin_amdgcn_wmma_f32_16x16x32_bf16(
                              false, afr[mt], false, bf,
                              (short)0, acc[mt], false, false);
            }
        }

        if (more) {
            wait_async0();        // own async copies landed
            __syncthreads();      // everyone's copies landed / readers done
            cur ^= 1;
        }
    }

    // ---------------- fused epilogue ----------------
    const int d = wave;
    #pragma unroll
    for (int mt = 0; mt < 4; ++mt) {
        #pragma unroll
        for (int v = 0; v < 8; ++v) {
            const int m = mt * 16 + v + 8 * half;
            const int o = o0 + m;
            float tv = acc[mt][v] * w_syn[((size_t)o * DD + d) * SS + sidx];
            tv += __shfl_xor(tv, 1);
            tv += __shfl_xor(tv, 2);
            tv += __shfl_xor(tv, 4);
            tv += __shfl_xor(tv, 8);
            if (sidx == 0) {
                const float dv = tanhf(tv + b_dend[o * DD + d]);
                partial[m][d] = dv * w_dend[o * DD + d];
            }
        }
    }
    __syncthreads();
    if (t < OT) {
        const int o = o0 + t;
        float s = b_soma[o];
        #pragma unroll
        for (int dd = 0; dd < DD; ++dd) s += partial[t][dd];
        out[(size_t)b * NN + o] = 1.0f / (1.0f + __expf(-s));
    }
}

// ===========================================================================
// Fallback: register-staged f32 inputs, no workspace.
// ===========================================================================
__global__ __launch_bounds__(256)
void snn_fused_fallback(const float* __restrict__ x, const float* __restrict__ conn,
                        const float* __restrict__ w_syn, const float* __restrict__ b_dend,
                        const float* __restrict__ w_dend, const float* __restrict__ b_soma,
                        float* __restrict__ out)
{
    __shared__ __bf16 As[OT * LSTRIDE];
    __shared__ __bf16 Bst[COLS * LSTRIDE];
    __shared__ float  partial[OT][DD];

    const int t = threadIdx.x, wave = t >> 5, lane = t & 31;
    const int half = lane >> 4, sidx = lane & 15;
    const int b = blockIdx.y, o0 = blockIdx.x * OT;
    const int xr0 = (t >> 5) * 2, xc0 = (t & 31) * 4;
    const int ar = t >> 4, ac0 = (t & 15) * 4;

    const float* xptr = x + ((size_t)(b * NN) + xr0) * COLS + xc0;
    const float* aptr = conn + (size_t)(o0 + ar) * NN + ac0;

    v8f acc[4] = {};
    v4f xreg[8]; v4f mreg[4];
    #pragma unroll
    for (int p = 0; p < 4; ++p) {
        xreg[2 * p]     = *(const v4f*)(xptr + (p * 16) * COLS);
        xreg[2 * p + 1] = *(const v4f*)(xptr + (p * 16 + 1) * COLS);
        mreg[p]         = *(const v4f*)(aptr + (size_t)(p * 16) * NN);
    }
    xptr += (size_t)KC * COLS; aptr += KC;

    for (int k0 = 0; k0 < NN; k0 += KC) {
        __syncthreads();
        #pragma unroll
        for (int p = 0; p < 4; ++p) {
            const int r = p * 16 + xr0;
            #pragma unroll
            for (int j = 0; j < 4; ++j) {
                v2bf pk; pk[0] = (__bf16)xreg[2 * p][j]; pk[1] = (__bf16)xreg[2 * p + 1][j];
                *(v2bf*)&Bst[(xc0 + j) * LSTRIDE + r] = pk;
            }
            v4bf mb;
            #pragma unroll
            for (int j = 0; j < 4; ++j) mb[j] = (__bf16)mreg[p][j];
            *(v4bf*)&As[(p * 16 + ar) * LSTRIDE + ac0] = mb;
        }
        __syncthreads();
        if (k0 + KC < NN) {
            #pragma unroll
            for (int p = 0; p < 4; ++p) {
                xreg[2 * p]     = *(const v4f*)(xptr + (p * 16) * COLS);
                xreg[2 * p + 1] = *(const v4f*)(xptr + (p * 16 + 1) * COLS);
                mreg[p]         = *(const v4f*)(aptr + (size_t)(p * 16) * NN);
            }
            xptr += (size_t)KC * COLS; aptr += KC;
        }
        #pragma unroll
        for (int kk = 0; kk < KC; kk += 32) {
            const __bf16* bp = &Bst[(wave * 16 + sidx) * LSTRIDE + kk + half * 16];
            v8bf b0 = *(const v8bf*)bp; v8bf b1 = *(const v8bf*)(bp + 8);
            v16bf bf;
            #pragma unroll
            for (int i = 0; i < 8; ++i) { bf[i] = b0[i]; bf[i + 8] = b1[i]; }
            #pragma unroll
            for (int mt = 0; mt < 4; ++mt) {
                const __bf16* ap = &As[(mt * 16 + sidx) * LSTRIDE + kk + half * 16];
                v8bf a0 = *(const v8bf*)ap; v8bf a1 = *(const v8bf*)(ap + 8);
                v16bf a;
                #pragma unroll
                for (int i = 0; i < 8; ++i) { a[i] = a0[i]; a[i + 8] = a1[i]; }
                acc[mt] = __builtin_amdgcn_wmma_f32_16x16x32_bf16(
                              false, a, false, bf, (short)0, acc[mt], false, false);
            }
        }
    }
    const int d = wave;
    #pragma unroll
    for (int mt = 0; mt < 4; ++mt) {
        #pragma unroll
        for (int v = 0; v < 8; ++v) {
            const int m = mt * 16 + v + 8 * half;
            const int o = o0 + m;
            float tv = acc[mt][v] * w_syn[((size_t)o * DD + d) * SS + sidx];
            tv += __shfl_xor(tv, 1); tv += __shfl_xor(tv, 2);
            tv += __shfl_xor(tv, 4); tv += __shfl_xor(tv, 8);
            if (sidx == 0) {
                const float dv = tanhf(tv + b_dend[o * DD + d]);
                partial[m][d] = dv * w_dend[o * DD + d];
            }
        }
    }
    __syncthreads();
    if (t < OT) {
        const int o = o0 + t;
        float s = b_soma[o];
        #pragma unroll
        for (int dd = 0; dd < DD; ++dd) s += partial[t][dd];
        out[(size_t)b * NN + o] = 1.0f / (1.0f + __expf(-s));
    }
}

extern "C" void kernel_launch(void* const* d_in, const int* in_sizes, int n_in,
                              void* d_out, int out_size, void* d_ws, size_t ws_size,
                              hipStream_t stream) {
    const float* x      = (const float*)d_in[0];
    const float* conn   = (const float*)d_in[1];
    const float* w_syn  = (const float*)d_in[2];
    const float* b_dend = (const float*)d_in[3];
    const float* w_dend = (const float*)d_in[4];
    const float* b_soma = (const float*)d_in[5];
    float* out = (float*)d_out;

    const size_t conn_bytes = (size_t)NN * NN * sizeof(__bf16);              // 2 MB
    const size_t xT_bytes   = (size_t)BB * COLS * NN * sizeof(__bf16);       // 16 MB

    if (ws_size >= conn_bytes + xT_bytes) {
        __bf16* connb = (__bf16*)d_ws;
        __bf16* xT    = (__bf16*)((char*)d_ws + conn_bytes);

        conn_to_bf16_kernel<<<dim3(NN * NN / (256 * 8)), dim3(256), 0, stream>>>(conn, connb);
        x_transpose_bf16_kernel<<<dim3(NN / KC, BB), dim3(256), 0, stream>>>(x, xT);
        snn_wmma_async_kernel<<<dim3(NN / OT, BB), dim3(256), 0, stream>>>(
            xT, connb, w_syn, b_dend, w_dend, b_soma, out);
    } else {
        snn_fused_fallback<<<dim3(NN / OT, BB), dim3(256), 0, stream>>>(
            x, conn, w_syn, b_dend, w_dend, b_soma, out);
    }
}
